// Transformer_16887811408555
// MI455X (gfx1250) — compile-verified, matrix operand order checked by reading, not run on hardware
//
#include <hip/hip_runtime.h>
#include <hip/hip_bf16.h>

// ---------------------------------------------------------------------------
// Problem constants (match reference)
// ---------------------------------------------------------------------------
#define BB   256
#define SS   128
#define IN_  7
#define DD   1000
#define HH   4
#define DHH  250
#define FF_  1000
#define OUT_ 1000
#define LL   4

typedef __attribute__((ext_vector_type(16))) __bf16         bf16x16;
typedef __attribute__((ext_vector_type(8)))  float          f32x8;
typedef __attribute__((ext_vector_type(8)))  unsigned short us8;
typedef __attribute__((ext_vector_type(4)))  int            v4i;

struct Frag { us8 lo; us8 hi; };   // 32B, bit-cast to bf16x16

__device__ __forceinline__ unsigned short f2bf(float f) {
    unsigned u = __builtin_bit_cast(unsigned, f);
    u += 0x7FFFu + ((u >> 16) & 1u);       // round-to-nearest-even
    return (unsigned short)(u >> 16);
}

// ---------------------------------------------------------------------------
// Optional CDNA5 async global->LDS staging (ASYNCcnt path), compile-safe
// ---------------------------------------------------------------------------
#if __has_builtin(__builtin_amdgcn_global_load_async_to_lds_b128) && \
    __has_builtin(__builtin_amdgcn_s_wait_asynccnt)
#define ASYNC_OK 1
#else
#define ASYNC_OK 0
#endif

#if ASYNC_OK
__device__ __forceinline__ void async_copy_b128(const unsigned short* g,
                                                unsigned short* l) {
    __builtin_amdgcn_global_load_async_to_lds_b128(
        (__attribute__((address_space(1))) v4i*)(unsigned long long)(size_t)g,
        (__attribute__((address_space(3))) v4i*)(unsigned)(size_t)l,
        0, 0);
}
#endif

// ---------------------------------------------------------------------------
// Generic bf16 WMMA GEMM:  C[M,N] = scale * A[M,K] * B + bias, opt ReLU
//   B_NK  : B stored [N,K] (weights) vs [K,N] (V matrix, transposed in LDS)
//   AL16  : A (and B if B_NK) rows are 16B aligned -> b128 staging / async
// Batched via gridDim.z: z -> (z/Hdiv, z%Hdiv) with (outer,inner) offsets.
// Block tile 128x128, 8 waves (4Mx2N), wave tile 32x64 = 2x4 WMMA 16x16x32.
// Double-buffered LDS, one barrier per K-step. Requires M % 128 == 0.
// ---------------------------------------------------------------------------
template <bool B_NK, bool AL16>
__global__ __launch_bounds__(256) void wmma_gemm(
    const unsigned short* __restrict__ A, const unsigned short* __restrict__ Bm,
    const float* __restrict__ bias, float* __restrict__ outF,
    unsigned short* __restrict__ outB,
    int M, int N, int K, int lda, int ldb, int ldc,
    float scale, int relu, int Hdiv,
    long long aO, long long aI, long long bO, long long bI,
    long long cO, long long cI)
{
    __shared__ __align__(16) unsigned short As[2][128 * 40];
    __shared__ __align__(16) unsigned short Bs[2][128 * 40];

    const int z  = blockIdx.z;
    const int zb = z / Hdiv, zh = z % Hdiv;
    A  += (size_t)zb * aO + (size_t)zh * aI;
    Bm += (size_t)zb * bO + (size_t)zh * bI;
    const size_t cbase = (size_t)zb * cO + (size_t)zh * cI;

    const int m0   = blockIdx.x * 128;
    const int n0   = blockIdx.y * 128;
    const int tid  = threadIdx.x;
    const int lane = tid & 31;
    const int wid  = tid >> 5;
    const int wm   = wid & 3;      // 4 waves along M (32 rows each)
    const int wn   = wid >> 2;     // 2 waves along N (64 cols each)
    const int half = lane >> 4;
    const int l16  = lane & 15;

    f32x8 acc[2][4];
    #pragma unroll
    for (int i = 0; i < 2; ++i)
        #pragma unroll
        for (int j = 0; j < 4; ++j)
            #pragma unroll
            for (int r = 0; r < 8; ++r) acc[i][j][r] = 0.0f;

    const int nkb   = (K + 31) >> 5;
    const int tailk = K & 31;

    // staging registers (register-pipelined path)
    int4 aR[2], bR[2];            // AL16: 2x b128 each
    unsigned aw[8], bw[8];        // !AL16 / strided-B: 8x b32 each

    auto loadA = [&](int kb) {
        const int k0 = kb << 5;
        const bool tail = (kb == nkb - 1) && tailk;
        if constexpr (AL16) {
            if (!tail) {
                #pragma unroll
                for (int i = 0; i < 2; ++i) {
                    int p = tid + i * 256, row = p >> 2, seg = p & 3;
                    aR[i] = *(const int4*)(A + (size_t)(m0 + row) * lda + k0 + seg * 8);
                }
            } else {
                #pragma unroll
                for (int i = 0; i < 2; ++i) {
                    int p = tid + i * 256, row = p >> 2, seg = p & 3;
                    const unsigned short* ap = A + (size_t)(m0 + row) * lda;
                    unsigned wv[4];
                    #pragma unroll
                    for (int j = 0; j < 4; ++j) {
                        int kk = k0 + seg * 8 + j * 2;
                        wv[j] = (kk + 1 < K) ? *(const unsigned*)(ap + kk)
                              : (kk < K ? (unsigned)ap[kk] : 0u);
                    }
                    aR[i] = make_int4(wv[0], wv[1], wv[2], wv[3]);
                }
            }
        } else {
            #pragma unroll
            for (int i = 0; i < 8; ++i) {
                int p = tid + i * 256, row = p >> 4, pc = p & 15;
                int kk = k0 + pc * 2;
                const unsigned short* ap = A + (size_t)(m0 + row) * lda;
                if (!tail)            aw[i] = *(const unsigned*)(ap + kk);
                else if (kk + 1 < K)  aw[i] = *(const unsigned*)(ap + kk);
                else if (kk < K)      aw[i] = (unsigned)ap[kk];
                else                  aw[i] = 0u;
            }
        }
    };
    auto storeA = [&](int c) {
        if constexpr (AL16) {
            #pragma unroll
            for (int i = 0; i < 2; ++i) {
                int p = tid + i * 256, row = p >> 2, seg = p & 3;
                *(int4*)(&As[c][row * 40 + seg * 8]) = aR[i];
            }
        } else {
            #pragma unroll
            for (int i = 0; i < 8; ++i) {
                int p = tid + i * 256, row = p >> 4, pc = p & 15;
                *(unsigned*)(&As[c][row * 40 + pc * 2]) = aw[i];
            }
        }
    };
    auto loadB = [&](int kb) {
        const int k0 = kb << 5;
        const bool tail = (kb == nkb - 1) && tailk;
        if constexpr (B_NK && AL16) {
            #pragma unroll
            for (int i = 0; i < 2; ++i) {
                int p = tid + i * 256, row = p >> 2, seg = p & 3;
                int gn = n0 + row;
                int4 v = make_int4(0, 0, 0, 0);
                if (gn < N) {
                    const unsigned short* bp = Bm + (size_t)gn * ldb;
                    if (!tail) v = *(const int4*)(bp + k0 + seg * 8);
                    else {
                        unsigned wv[4];
                        #pragma unroll
                        for (int j = 0; j < 4; ++j) {
                            int kk = k0 + seg * 8 + j * 2;
                            wv[j] = (kk + 1 < K) ? *(const unsigned*)(bp + kk)
                                  : (kk < K ? (unsigned)bp[kk] : 0u);
                        }
                        v = make_int4(wv[0], wv[1], wv[2], wv[3]);
                    }
                }
                bR[i] = v;
            }
        } else if constexpr (B_NK) {
            #pragma unroll
            for (int i = 0; i < 8; ++i) {
                int p = tid + i * 256, row = p >> 4, pc = p & 15;
                int kk = k0 + pc * 2, gn = n0 + row;
                unsigned v = 0;
                if (gn < N) {
                    const unsigned short* bp = Bm + (size_t)gn * ldb;
                    if (!tail)           v = *(const unsigned*)(bp + kk);
                    else if (kk + 1 < K) v = *(const unsigned*)(bp + kk);
                    else if (kk < K)     v = (unsigned)bp[kk];
                }
                bw[i] = v;
            }
        } else {
            // B stored [K,N]: transpose-gather into [n][k] tile
            #pragma unroll
            for (int i = 0; i < 8; ++i) {
                int p = tid + i * 256, row = p >> 4, pc = p & 15;
                int kk = k0 + pc * 2, gn = n0 + row;
                unsigned short e0 = 0, e1 = 0;
                if (gn < N) {
                    if (kk     < K) e0 = Bm[(size_t)(kk    ) * ldb + gn];
                    if (kk + 1 < K) e1 = Bm[(size_t)(kk + 1) * ldb + gn];
                }
                bw[i] = (unsigned)e0 | ((unsigned)e1 << 16);
            }
        }
    };
    auto storeB = [&](int c) {
        if constexpr (B_NK && AL16) {
            #pragma unroll
            for (int i = 0; i < 2; ++i) {
                int p = tid + i * 256, row = p >> 2, seg = p & 3;
                *(int4*)(&Bs[c][row * 40 + seg * 8]) = bR[i];
            }
        } else {
            #pragma unroll
            for (int i = 0; i < 8; ++i) {
                int p = tid + i * 256, row = p >> 4, pc = p & 15;
                *(unsigned*)(&Bs[c][row * 40 + pc * 2]) = bw[i];
            }
        }
    };

    auto compute = [&](int c) {
        Frag af[2], bfr[4];
        #pragma unroll
        for (int mi = 0; mi < 2; ++mi) {
            int row = wm * 32 + mi * 16 + l16;
            af[mi].lo = *(const us8*)(&As[c][row * 40 + half * 8]);
            af[mi].hi = *(const us8*)(&As[c][row * 40 + half * 8 + 16]);
        }
        #pragma unroll
        for (int ni = 0; ni < 4; ++ni) {
            int row = wn * 64 + ni * 16 + l16;
            bfr[ni].lo = *(const us8*)(&Bs[c][row * 40 + half * 8]);
            bfr[ni].hi = *(const us8*)(&Bs[c][row * 40 + half * 8 + 16]);
        }
        #pragma unroll
        for (int mi = 0; mi < 2; ++mi)
            #pragma unroll
            for (int ni = 0; ni < 4; ++ni)
                acc[mi][ni] = __builtin_amdgcn_wmma_f32_16x16x32_bf16(
                    false, __builtin_bit_cast(bf16x16, af[mi]),
                    false, __builtin_bit_cast(bf16x16, bfr[ni]),
                    (short)0, acc[mi][ni], false, false);
    };

#if ASYNC_OK
    if constexpr (AL16 && B_NK) {
        // ---- async global->LDS pipeline (ASYNCcnt path), 1 barrier/step
        auto stage_async = [&](int kb, int c) {
            const int k0 = kb << 5;
            if ((kb == nkb - 1) && tailk) {          // guarded tail via regs
                loadA(kb); loadB(kb); storeA(c); storeB(c);
            } else {
                #pragma unroll
                for (int i = 0; i < 2; ++i) {
                    int p = tid + i * 256, row = p >> 2, seg = p & 3;
                    async_copy_b128(A + (size_t)(m0 + row) * lda + k0 + seg * 8,
                                    &As[c][row * 40 + seg * 8]);
                }
                #pragma unroll
                for (int i = 0; i < 2; ++i) {
                    int p = tid + i * 256, row = p >> 2, seg = p & 3;
                    int gn = n0 + row;
                    if (gn < N)   // out-of-range cols never leave the epilogue
                        async_copy_b128(Bm + (size_t)gn * ldb + k0 + seg * 8,
                                        &Bs[c][row * 40 + seg * 8]);
                }
            }
        };
        stage_async(0, 0);
        __builtin_amdgcn_s_wait_asynccnt(0);
        __syncthreads();
        for (int kb = 0; kb < nkb; ++kb) {
            int c = kb & 1;
            if (kb + 1 < nkb) stage_async(kb + 1, c ^ 1);
            compute(c);
            __builtin_amdgcn_s_wait_asynccnt(0);
            __syncthreads();
        }
    } else
#endif
    {
        // ---- register-pipelined double buffer, 1 barrier/step
        loadA(0); loadB(0);
        for (int kb = 0; kb < nkb; ++kb) {
            int c = kb & 1;
            storeA(c); storeB(c);
            __syncthreads();
            if (kb + 1 < nkb) { loadA(kb + 1); loadB(kb + 1); }
            compute(c);
        }
    }

    // ---- epilogue: scale, bias, relu, dual-precision store ----
    #pragma unroll
    for (int mi = 0; mi < 2; ++mi) {
        #pragma unroll
        for (int ni = 0; ni < 4; ++ni) {
            int gn = n0 + wn * 64 + ni * 16 + l16;
            if (gn >= N) continue;
            float bval = bias ? bias[gn] : 0.0f;
            #pragma unroll
            for (int r = 0; r < 8; ++r) {
                int gm = m0 + wm * 32 + mi * 16 + half * 8 + r;
                float v = acc[mi][ni][r] * scale + bval;
                if (relu) v = v > 0.0f ? v : 0.0f;
                size_t idx = cbase + (size_t)gm * ldc + gn;
                if (outF) outF[idx] = v;
                if (outB) outB[idx] = f2bf(v);
            }
        }
    }
}

// ---------------------------------------------------------------------------
// Embedding: h[t,d] = sum_i x[t,i]*We[d,i] + be[d]   (K=7 scalar)
// ---------------------------------------------------------------------------
__global__ __launch_bounds__(256) void embed_kernel(
    const float* __restrict__ x, const float* __restrict__ We,
    const float* __restrict__ be, float* __restrict__ hF,
    unsigned short* __restrict__ hB)
{
    int idx = blockIdx.x * 256 + threadIdx.x;
    if (idx >= BB * SS * DD) return;
    int t = idx / DD, d = idx % DD;
    const float* xr = x + (size_t)t * IN_;
    const float* wr = We + (size_t)d * IN_;
    float s = be[d];
    #pragma unroll
    for (int i = 0; i < IN_; ++i) s += xr[i] * wr[i];
    hF[idx] = s;
    hB[idx] = f2bf(s);
}

// ---------------------------------------------------------------------------
// Masked softmax over k (row length 128), one wave per row, bf16 output
// ---------------------------------------------------------------------------
__global__ __launch_bounds__(256) void softmax_kernel(
    const float* __restrict__ scores, unsigned short* __restrict__ probs,
    const int* __restrict__ len)
{
    int row  = blockIdx.x * 8 + (threadIdx.x >> 5);
    int lane = threadIdx.x & 31;
    int b = row / (HH * SS);
    int Lv = len[b];
    const float* sr = scores + (size_t)row * SS;
    float x[4];
    #pragma unroll
    for (int j = 0; j < 4; ++j) {
        int k = lane + 32 * j;
        x[j] = (k < Lv) ? sr[k] : -1e9f;
    }
    float m = fmaxf(fmaxf(x[0], x[1]), fmaxf(x[2], x[3]));
    #pragma unroll
    for (int off = 16; off > 0; off >>= 1) m = fmaxf(m, __shfl_xor(m, off, 32));
    float s = 0.0f;
    #pragma unroll
    for (int j = 0; j < 4; ++j) { x[j] = __expf(x[j] - m); s += x[j]; }
    #pragma unroll
    for (int off = 16; off > 0; off >>= 1) s += __shfl_xor(s, off, 32);
    float inv = 1.0f / s;
    unsigned short* pr = probs + (size_t)row * SS;
    #pragma unroll
    for (int j = 0; j < 4; ++j) pr[lane + 32 * j] = f2bf(x[j] * inv);
}

// ---------------------------------------------------------------------------
// Residual add + LayerNorm, one block per token, writes f32 + bf16
// ---------------------------------------------------------------------------
__device__ __forceinline__ float block_sum(float v, float* red) {
    #pragma unroll
    for (int off = 16; off > 0; off >>= 1) v += __shfl_xor(v, off, 32);
    int lane = threadIdx.x & 31, wid = threadIdx.x >> 5;
    if (lane == 0) red[wid] = v;
    __syncthreads();
    float t = 0.0f;
    #pragma unroll
    for (int i = 0; i < 8; ++i) t += red[i];
    __syncthreads();
    return t;
}

__global__ __launch_bounds__(256) void add_ln_kernel(
    const float* __restrict__ resid, const float* __restrict__ delta,
    const float* __restrict__ g, const float* __restrict__ beta,
    float* __restrict__ outF, unsigned short* __restrict__ outB)
{
    __shared__ float red[8];
    size_t base = (size_t)blockIdx.x * DD;
    float x[4];
    float s = 0.0f;
    #pragma unroll
    for (int j = 0; j < 4; ++j) {
        int i = threadIdx.x + 256 * j;
        x[j] = (i < DD) ? (resid[base + i] + delta[base + i]) : 0.0f;
        s += x[j];
    }
    float mean = block_sum(s, red) * (1.0f / DD);
    float vs = 0.0f;
    #pragma unroll
    for (int j = 0; j < 4; ++j) {
        int i = threadIdx.x + 256 * j;
        if (i < DD) { float dv = x[j] - mean; vs += dv * dv; }
    }
    float var = block_sum(vs, red) * (1.0f / DD);
    float rstd = rsqrtf(var + 1e-5f);
    #pragma unroll
    for (int j = 0; j < 4; ++j) {
        int i = threadIdx.x + 256 * j;
        if (i < DD) {
            float y = (x[j] - mean) * rstd * g[i] + beta[i];
            outF[base + i] = y;
            outB[base + i] = f2bf(y);
        }
    }
}

// ---------------------------------------------------------------------------
// Masked sum-pool over tokens + ReLU, bf16 out (head input)
// ---------------------------------------------------------------------------
__global__ __launch_bounds__(256) void pool_kernel(
    const float* __restrict__ h, const int* __restrict__ len,
    unsigned short* __restrict__ out)
{
    int idx = blockIdx.x * 256 + threadIdx.x;
    if (idx >= BB * DD) return;
    int b = idx / DD, d = idx % DD;
    int Lv = len[b];
    float s = 0.0f;
    for (int t = 0; t < Lv; ++t)
        s += h[((size_t)b * SS + t) * DD + d];
    out[idx] = f2bf(fmaxf(s, 0.0f));
}

// ---------------------------------------------------------------------------
// fp32 -> bf16 elementwise cast (weights)
// ---------------------------------------------------------------------------
__global__ __launch_bounds__(256) void cast_bf16_kernel(
    const float* __restrict__ in, unsigned short* __restrict__ out, int n)
{
    int idx = blockIdx.x * 256 + threadIdx.x;
    if (idx < n) out[idx] = f2bf(in[idx]);
}

// ---------------------------------------------------------------------------
// Host driver
// ---------------------------------------------------------------------------
extern "C" void kernel_launch(void* const* d_in, const int* in_sizes, int n_in,
                              void* d_out, int out_size, void* d_ws, size_t ws_size,
                              hipStream_t stream)
{
    (void)in_sizes; (void)n_in; (void)out_size; (void)ws_size;

    const float* view[2] = { (const float*)d_in[0], (const float*)d_in[1] };
    const int*   lens[2] = { (const int*)d_in[2],   (const int*)d_in[3]   };
    const float* We   = (const float*)d_in[4];
    const float* be   = (const float*)d_in[5];
    const float* Wqkv = (const float*)d_in[6];
    const float* bqkv = (const float*)d_in[7];
    const float* Wo   = (const float*)d_in[8];
    const float* bo   = (const float*)d_in[9];
    const float* g1   = (const float*)d_in[10];
    const float* be1  = (const float*)d_in[11];
    const float* W1   = (const float*)d_in[12];
    const float* b1   = (const float*)d_in[13];
    const float* W2   = (const float*)d_in[14];
    const float* b2   = (const float*)d_in[15];
    const float* g2   = (const float*)d_in[16];
    const float* be2  = (const float*)d_in[17];
    const float* Wh1  = (const float*)d_in[18];
    const float* bh1  = (const float*)d_in[19];
    const float* Wh2  = (const float*)d_in[20];
    const float* bh2  = (const float*)d_in[21];
    float* out = (float*)d_out;

    // ---- workspace carve-up ----
    char* w = (char*)d_ws;
    size_t off = 0;
    auto carve = [&](size_t bytes) -> void* {
        void* p = w + off;
        off += (bytes + 255) & ~(size_t)255;
        return p;
    };
    unsigned short* wqkv_bf = (unsigned short*)carve((size_t)LL * 3 * DD * DD * 2);
    unsigned short* wo_bf   = (unsigned short*)carve((size_t)LL * DD * DD * 2);
    unsigned short* w1_bf   = (unsigned short*)carve((size_t)LL * FF_ * DD * 2);
    unsigned short* w2_bf   = (unsigned short*)carve((size_t)LL * DD * FF_ * 2);
    unsigned short* wh1_bf  = (unsigned short*)carve((size_t)OUT_ * DD * 2);
    unsigned short* wh2_bf  = (unsigned short*)carve((size_t)OUT_ * OUT_ * 2);
    float*          h_f     = (float*)carve((size_t)BB * SS * DD * 4);
    unsigned short* h_bf    = (unsigned short*)carve((size_t)BB * SS * DD * 2);
    unsigned short* qkv_bf  = (unsigned short*)carve((size_t)BB * SS * 3 * DD * 2);
    float*          scor_f  = (float*)carve((size_t)BB * HH * SS * SS * 4);
    unsigned short* prob_bf = (unsigned short*)carve((size_t)BB * HH * SS * SS * 2);
    unsigned short* attn_bf = (unsigned short*)carve((size_t)BB * SS * DD * 2);
    float*          tmp_f   = (float*)carve((size_t)BB * SS * DD * 4);
    unsigned short* ff_bf   = (unsigned short*)carve((size_t)BB * SS * FF_ * 2);
    unsigned short* pool_bf = (unsigned short*)carve((size_t)BB * DD * 2);
    unsigned short* z1_bf   = (unsigned short*)carve((size_t)BB * OUT_ * 2);

    auto cast = [&](const float* src, unsigned short* dst, int n) {
        cast_bf16_kernel<<<(n + 255) / 256, 256, 0, stream>>>(src, dst, n);
    };
    cast(Wqkv, wqkv_bf, LL * 3 * DD * DD);
    cast(Wo,   wo_bf,   LL * DD * DD);
    cast(W1,   w1_bf,   LL * FF_ * DD);
    cast(W2,   w2_bf,   LL * DD * FF_);
    cast(Wh1,  wh1_bf,  OUT_ * DD);
    cast(Wh2,  wh2_bf,  OUT_ * OUT_);

    // plain (non-batched) GEMM against [N,K] weights, 16B-aligned rows
    auto gemm_nk = [&](const unsigned short* A, const unsigned short* Wm,
                       const float* bias, float* oF, unsigned short* oB,
                       int M, int N, int K, int lda, int ldb, int ldc,
                       int relu) {
        dim3 g(M / 128, (N + 127) / 128, 1);
        wmma_gemm<true, true><<<g, 256, 0, stream>>>(A, Wm, bias, oF, oB,
            M, N, K, lda, ldb, ldc, 1.0f, relu, 1, 0, 0, 0, 0, 0, 0);
    };

    const int M = BB * SS;                    // 32768 tokens
    const float inv_sqrt_dh = 0.0632455532f;  // 1/sqrt(250)

    for (int v = 0; v < 2; ++v) {
        // --- embedding ---
        embed_kernel<<<(M * DD + 255) / 256, 256, 0, stream>>>(
            view[v], We, be, h_f, h_bf);

        for (int l = 0; l < LL; ++l) {
            // --- QKV projection: [32768,1000] x [3000,1000]^T -> bf16 ---
            gemm_nk(h_bf, wqkv_bf + (size_t)l * 3 * DD * DD, bqkv + l * 3 * DD,
                    nullptr, qkv_bf, M, 3 * DD, DD, DD, DD, 3 * DD, 0);

            // --- scores = Q K^T / sqrt(DH), batched over (b,h) ---
            {
                dim3 g(SS / 128, 1, BB * HH);
                wmma_gemm<true, false><<<g, 256, 0, stream>>>(
                    qkv_bf,                 // Q base
                    qkv_bf + DD,            // K base
                    nullptr, scor_f, nullptr,
                    SS, SS, DHH, 3 * DD, 3 * DD, SS,
                    inv_sqrt_dh, 0, HH,
                    (long long)SS * 3 * DD, DHH,                  // A: b,h offs
                    (long long)SS * 3 * DD, DHH,                  // B: b,h offs
                    (long long)HH * SS * SS, (long long)SS * SS); // C
            }
            // --- masked softmax -> bf16 probs ---
            softmax_kernel<<<BB * HH * SS / 8, 256, 0, stream>>>(
                scor_f, prob_bf, lens[v]);

            // --- A = P V  (V is [K,N] layout), batched ---
            {
                dim3 g(SS / 128, (DHH + 127) / 128, BB * HH);
                wmma_gemm<false, true><<<g, 256, 0, stream>>>(
                    prob_bf,
                    qkv_bf + 2 * DD,        // V base
                    nullptr, nullptr, attn_bf,
                    SS, DHH, SS, SS, 3 * DD, DD,
                    1.0f, 0, HH,
                    (long long)HH * SS * SS, (long long)SS * SS,  // A
                    (long long)SS * 3 * DD, DHH,                  // B
                    (long long)SS * DD, DHH);                     // C -> [B,S,h*DH+d]
            }
            // --- output projection + residual LN ---
            gemm_nk(attn_bf, wo_bf + (size_t)l * DD * DD, bo + l * DD,
                    tmp_f, nullptr, M, DD, DD, DD, DD, DD, 0);
            add_ln_kernel<<<M, 256, 0, stream>>>(
                h_f, tmp_f, g1 + l * DD, be1 + l * DD, h_f, h_bf);

            // --- feedforward ---
            gemm_nk(h_bf, w1_bf + (size_t)l * FF_ * DD, b1 + l * FF_,
                    nullptr, ff_bf, M, FF_, DD, DD, DD, FF_, 1);
            gemm_nk(ff_bf, w2_bf + (size_t)l * DD * FF_, b2 + l * DD,
                    tmp_f, nullptr, M, DD, FF_, FF_, FF_, DD, 0);
            add_ln_kernel<<<M, 256, 0, stream>>>(
                h_f, tmp_f, g2 + l * DD, be2 + l * DD, h_f, h_bf);
        }

        // --- masked sum pool + ReLU -> head input ---
        pool_kernel<<<(BB * DD + 255) / 256, 256, 0, stream>>>(h_f, lens[v], pool_bf);

        // --- head: linear(relu) -> linear, f32 out into d_out slice ---
        gemm_nk(pool_bf, wh1_bf, bh1, nullptr, z1_bf,
                BB, OUT_, DD, DD, DD, OUT_, 1);
        gemm_nk(z1_bf, wh2_bf, bh2, out + (size_t)v * BB * OUT_, nullptr,
                BB, OUT_, OUT_, OUT_, OUT_, OUT_, 0);
    }
}
